// MultiHeadAttention_49821620633811
// MI455X (gfx1250) — matrix operand order repaired and checked by reading, not compiled, not run on hardware
//
#include <hip/hip_runtime.h>
#include <hip/hip_bf16.h>

typedef __attribute__((ext_vector_type(16))) _Float16 v16h;
typedef __attribute__((ext_vector_type(8)))  _Float16 v8h;
typedef __attribute__((ext_vector_type(8)))  float    v8f;
typedef int v4i __attribute__((vector_size(4 * sizeof(int))));

#define D_MODELC 1024
#define N_HEADSC 16
#define N_KVC    4
#define D_KC     64
#define Q_SIZEC  1024
#define KV_SIZEC 256
#define QKV_SIZEC 1536
#define BATCHC   2
#define SEQC     2048
#define MTOKC    (BATCHC*SEQC)   /* 4096 */
#define EPSC     1.1920929e-07f
#define NEG_BIG  (-1.0e30f)

// ---- CDNA5 async global->LDS path (guarded; falls back to sync staging) ----
#if defined(__has_builtin)
#if __has_builtin(__builtin_amdgcn_global_load_async_to_lds_b128) && \
    __has_builtin(__builtin_amdgcn_s_wait_asynccnt)
#define USE_ASYNC_LDS 1
#endif
#endif

#if defined(USE_ASYNC_LDS)
static __device__ __forceinline__ void async_ld16(const _Float16* g, _Float16* l) {
  // 16 bytes per lane, global -> LDS direct, tracked by ASYNCcnt
  __builtin_amdgcn_global_load_async_to_lds_b128(
      (__attribute__((address_space(1))) v4i*)g,
      (__attribute__((address_space(3))) v4i*)l, 0, 0);
}
#endif

static __device__ __forceinline__ v8f v8f_zero() {
  v8f z = {0.f,0.f,0.f,0.f,0.f,0.f,0.f,0.f};
  return z;
}

// ---------------------------------------------------------------------------
// fp32 -> f16 pack
// ---------------------------------------------------------------------------
__global__ __launch_bounds__(256) void cvt_f32_to_f16(
    const float* __restrict__ in, _Float16* __restrict__ out, int n) {
  int i = blockIdx.x * 256 + threadIdx.x;
  if (i < n) out[i] = (_Float16)in[i];
}

// ---------------------------------------------------------------------------
// WMMA GEMM: C[M,N] = A[M,K] * W[N,K]^T ; A,W f16 row-major, C f32.
// Block tile 128x128, 8 waves (2x4), wave tile 64x32 = 4x2 tiles of 16x16.
// Double-buffered LDS staging (async global->LDS when available).
// M,N multiples of 128; K multiple of 32.
// ---------------------------------------------------------------------------
__global__ __launch_bounds__(256) void gemm_f16_wmma(
    const _Float16* __restrict__ A, const _Float16* __restrict__ W,
    float* __restrict__ C, int M, int N, int K)
{
  __shared__ __align__(64) _Float16 sA[2][128 * 32];
  __shared__ __align__(64) _Float16 sB[2][128 * 32];

  const int tid  = threadIdx.x;
  const int wave = tid >> 5;
  const int lane = tid & 31;
  const int l16  = lane & 15;
  const int lhi  = lane >> 4;           // 0: lanes 0-15, 1: lanes 16-31
  const int wm   = (wave >> 2) * 64;    // wave row base in block tile
  const int wn   = (wave & 3) * 32;     // wave col base in block tile
  const long bm0 = (long)blockIdx.y * 128;
  const long bn0 = (long)blockIdx.x * 128;

  v8f acc[4][2];
#pragma unroll
  for (int i = 0; i < 4; i++)
#pragma unroll
    for (int j = 0; j < 2; j++) acc[i][j] = v8f_zero();

  const int srow = tid >> 1;            // 0..127
  const int scol = (tid & 1) * 16;      // 0 or 16 (halfs)
  const _Float16* gA = A + (bm0 + srow) * (long)K + scol;
  const _Float16* gW = W + (bn0 + srow) * (long)K + scol;
  const int ntk = K >> 5;               // 32-wide k tiles

#if defined(USE_ASYNC_LDS)
  auto stage_async = [&](int buf, int k0) {
    const _Float16* ga = gA + k0;
    const _Float16* gw = gW + k0;
    _Float16* la = &sA[buf][srow * 32 + scol];
    _Float16* lb = &sB[buf][srow * 32 + scol];
    async_ld16(ga,     la);
    async_ld16(ga + 8, la + 8);
    async_ld16(gw,     lb);
    async_ld16(gw + 8, lb + 8);          // 4 async issues per wave per tile
  };
  stage_async(0, 0);                     // prologue: tile 0 -> buf 0
#endif

  for (int j = 0; j < ntk; j++) {
    const int cur = j & 1;
    const int k0  = j << 5;
#if defined(USE_ASYNC_LDS)
    if (j + 1 < ntk) {
      stage_async(cur ^ 1, k0 + 32);     // overlap next tile with compute
      __builtin_amdgcn_s_wait_asynccnt(4);
    } else {
      __builtin_amdgcn_s_wait_asynccnt(0);
    }
#else
    *(v16h*)&sA[cur][srow * 32 + scol] = *(const v16h*)(gA + k0);
    *(v16h*)&sB[cur][srow * 32 + scol] = *(const v16h*)(gW + k0);
    if (k0 + 32 < K) {
      __builtin_prefetch(gA + k0 + 32, 0, 1);   // global_prefetch_b8
      __builtin_prefetch(gW + k0 + 32, 0, 1);
    }
#endif
    __syncthreads();

    v16h a[4], b[2];
    // A fragment per ISA 16-bit A layout: lanes0-15 K{0..7,16..23}, lanes16-31 K{8..15,24..31}
#pragma unroll
    for (int i = 0; i < 4; i++) {
      const _Float16* p = &sA[cur][(wm + i * 16 + l16) * 32 + lhi * 8];
      v8h lo = *(const v8h*)p;
      v8h hi = *(const v8h*)(p + 16);
#pragma unroll
      for (int e = 0; e < 8; e++) { a[i][e] = lo[e]; a[i][e + 8] = hi[e]; }
    }
    // B fragment: lane n=l16, lanes0-15 K{0..7,16..23}, lanes16-31 K{8..15,24..31} (same as A)
#pragma unroll
    for (int jj = 0; jj < 2; jj++) {
      const _Float16* q = &sB[cur][(wn + jj * 16 + l16) * 32 + lhi * 8];
      v8h blo = *(const v8h*)q;
      v8h bhi = *(const v8h*)(q + 16);
#pragma unroll
      for (int e = 0; e < 8; e++) { b[jj][e] = blo[e]; b[jj][e + 8] = bhi[e]; }
    }

#pragma unroll
    for (int i = 0; i < 4; i++)
#pragma unroll
      for (int jj = 0; jj < 2; jj++)
        acc[i][jj] = __builtin_amdgcn_wmma_f32_16x16x32_f16(
            false, a[i], false, b[jj], (short)0, acc[i][jj], false, false);

    __syncthreads();
  }

  // C/D layout: VGPR r -> row r (lanes 0-15) / row 8+r (lanes 16-31), col = lane&15
#pragma unroll
  for (int i = 0; i < 4; i++)
#pragma unroll
    for (int j = 0; j < 2; j++)
#pragma unroll
      for (int r = 0; r < 8; r++) {
        long row = bm0 + wm + i * 16 + (lhi ? 8 + r : r);
        long col = bn0 + wn + j * 16 + l16;
        C[row * (long)N + col] = acc[i][j][r];
      }
}

// ---------------------------------------------------------------------------
// RMSNorm + RoPE. One wave per (b,t,head)-row.
// qkv f32 [B,T,1536] -> Qh f16 [B,H,T,64], Kh f16 [B,KV,T,64], Vt f16 [B,KV,64,T]
// ---------------------------------------------------------------------------
__global__ __launch_bounds__(256) void norm_rope_kernel(
    const float* __restrict__ qkv, const float* __restrict__ qw,
    const float* __restrict__ kw, _Float16* __restrict__ Qh,
    _Float16* __restrict__ Kh, _Float16* __restrict__ Vt)
{
  const int wave = threadIdx.x >> 5;
  const int lane = threadIdx.x & 31;   // RoPE pair index p: dims 2p, 2p+1
  int row = blockIdx.x * 8 + wave;

  const int QR = BATCHC * SEQC * N_HEADSC;   // 65536
  const int KR = BATCHC * SEQC * N_KVC;      // 16384
  int kind;
  if (row < QR)            { kind = 0; }
  else if (row < QR + KR)  { kind = 1; row -= QR; }
  else                     { kind = 2; row -= QR + KR; }

  const int nh = (kind == 0) ? N_HEADSC : N_KVC;
  const int h = row % nh;
  const int t = (row / nh) % SEQC;
  const int b = row / (nh * SEQC);

  const int coff = (kind == 0) ? h * D_KC
                 : (kind == 1) ? Q_SIZEC + h * D_KC
                               : Q_SIZEC + KV_SIZEC + h * D_KC;
  const float* src = qkv + ((long)(b * SEQC + t)) * QKV_SIZEC + coff;
  float x1 = src[2 * lane];
  float x2 = src[2 * lane + 1];

  if (kind < 2) {
    float ss = x1 * x1 + x2 * x2;
#pragma unroll
    for (int m = 16; m >= 1; m >>= 1) ss += __shfl_xor(ss, m, 32);
    float rr = rsqrtf(ss * (1.0f / 64.0f) + EPSC);
    const float* w = (kind == 0) ? qw : kw;
    x1 *= rr * w[2 * lane];
    x2 *= rr * w[2 * lane + 1];
    float inv_freq = powf(10000.0f, -((float)(2 * lane)) * (1.0f / 64.0f));
    float ang = (float)t * inv_freq;
    float sn, cs;
    sincosf(ang, &sn, &cs);
    float y1 = x1 * cs - x2 * sn;
    float y2 = x1 * sn + x2 * cs;
    if (kind == 0) {
      _Float16* d = Qh + (((long)(b * N_HEADSC + h)) * SEQC + t) * D_KC;
      d[2 * lane] = (_Float16)y1;
      d[2 * lane + 1] = (_Float16)y2;
    } else {
      _Float16* d = Kh + (((long)(b * N_KVC + h)) * SEQC + t) * D_KC;
      d[2 * lane] = (_Float16)y1;
      d[2 * lane + 1] = (_Float16)y2;
    }
  } else {
    // V: plain convert, store transposed [B,KV,64,T]
    _Float16* d = Vt + ((long)(b * N_KVC + h)) * D_KC * SEQC;
    d[(2 * lane) * SEQC + t]     = (_Float16)x1;
    d[(2 * lane + 1) * SEQC + t] = (_Float16)x2;
  }
}

// ---------------------------------------------------------------------------
// Causal flash attention with WMMA. Block = 128 queries of one (b,h);
// 8 waves x 16 query rows each; key blocks of 64; online softmax.
// Double-buffered async K/V staging. Oh f16 [B,T,1024]
// ---------------------------------------------------------------------------
__global__ __launch_bounds__(256) void flash_attn_wmma(
    const _Float16* __restrict__ Qh, const _Float16* __restrict__ Kh,
    const _Float16* __restrict__ Vt, _Float16* __restrict__ Oh)
{
  __shared__ __align__(64) _Float16 sK[2][64 * 64];    // [key][dim]
  __shared__ __align__(64) _Float16 sV[2][64 * 64];    // [dim][key]
  __shared__ __align__(64) _Float16 sP[8 * 16 * 64];   // per-wave P strip

  const int tid  = threadIdx.x;
  const int wave = tid >> 5;
  const int lane = tid & 31;
  const int l16  = lane & 15;
  const int lhi  = lane >> 4;
  const int bh = blockIdx.y;
  const int b  = bh / N_HEADSC;
  const int h  = bh % N_HEADSC;
  const int kv = h / (N_HEADSC / N_KVC);
  const int q0 = blockIdx.x * 128 + wave * 16;   // wave's first query token

  const _Float16* Qp = Qh + ((long)(b * N_HEADSC + h)) * SEQC * D_KC;
  const _Float16* Kp = Kh + ((long)(b * N_KVC + kv)) * SEQC * D_KC;
  const _Float16* Vp = Vt + ((long)(b * N_KVC + kv)) * D_KC * SEQC;

  // Load Q A-fragments (two K-chunks of 32 over D_K=64), kept in registers.
  v16h qa[2];
  {
    const _Float16* qrow = Qp + (long)(q0 + l16) * D_KC;
#pragma unroll
    for (int f = 0; f < 2; f++) {
      v8h lo = *(const v8h*)(qrow + f * 32 + lhi * 8);
      v8h hi = *(const v8h*)(qrow + f * 32 + lhi * 8 + 16);
#pragma unroll
      for (int e = 0; e < 8; e++) { qa[f][e] = lo[e]; qa[f][e + 8] = hi[e]; }
    }
  }

  v8f o[4];
  float m_i[8], l_i[8];
#pragma unroll
  for (int j = 0; j < 4; j++) o[j] = v8f_zero();
#pragma unroll
  for (int r = 0; r < 8; r++) { m_i[r] = NEG_BIG; l_i[r] = 0.f; }

  const int srow = tid >> 2;          // 0..63
  const int scol = (tid & 3) * 16;    // 0,16,32,48
  const int nkb = 2 * blockIdx.x + 2; // causal: keys up to end of q-block
  const float scale = 0.125f;         // 1/sqrt(64)

#if defined(USE_ASYNC_LDS)
  auto stage_async = [&](int buf, int kb) {
    const _Float16* gk = Kp + (long)(kb + srow) * D_KC + scol;
    const _Float16* gv = Vp + (long)srow * SEQC + kb + scol;
    _Float16* lk = &sK[buf][srow * 64 + scol];
    _Float16* lv = &sV[buf][srow * 64 + scol];
    async_ld16(gk,     lk);
    async_ld16(gk + 8, lk + 8);
    async_ld16(gv,     lv);
    async_ld16(gv + 8, lv + 8);          // 4 async issues per wave per tile
  };
  stage_async(0, 0);
#endif

  for (int jb = 0; jb < nkb; jb++) {
    const int kb0 = jb * 64;
    const int cur = jb & 1;
#if defined(USE_ASYNC_LDS)
    if (jb + 1 < nkb) {
      stage_async(cur ^ 1, kb0 + 64);
      __builtin_amdgcn_s_wait_asynccnt(4);
    } else {
      __builtin_amdgcn_s_wait_asynccnt(0);
    }
#else
    *(v16h*)&sK[cur][srow * 64 + scol] = *(const v16h*)(Kp + (long)(kb0 + srow) * D_KC + scol);
    *(v16h*)&sV[cur][srow * 64 + scol] = *(const v16h*)(Vp + (long)srow * SEQC + kb0 + scol);
#endif
    __syncthreads();

    if (kb0 <= q0 + 15) {   // skip fully-masked tiles for this wave
      // S = Q * K^T  (16 x 64 strip, 4 N-tiles)
      v8f csr[4];
#pragma unroll
      for (int j = 0; j < 4; j++) csr[j] = v8f_zero();
#pragma unroll
      for (int f = 0; f < 2; f++) {
#pragma unroll
        for (int j = 0; j < 4; j++) {
          v16h bk;
          {
            const _Float16* q = &sK[cur][(j * 16 + l16) * 64 + f * 32 + lhi * 8];
            v8h blo = *(const v8h*)q;
            v8h bhi = *(const v8h*)(q + 16);
#pragma unroll
            for (int e = 0; e < 8; e++) { bk[e] = blo[e]; bk[e + 8] = bhi[e]; }
          }
          csr[j] = __builtin_amdgcn_wmma_f32_16x16x32_f16(
              false, qa[f], false, bk, (short)0, csr[j], false, false);
        }
      }

      // scale + causal mask + online softmax (rows live on VGPR index,
      // spread over 16-lane half -> xor{8,4,2,1} reductions stay in-half)
      float rm[8], rs[8];
#pragma unroll
      for (int r = 0; r < 8; r++) {
        rm[r] = NEG_BIG;
        const int q = q0 + (lhi ? 8 + r : r);
#pragma unroll
        for (int j = 0; j < 4; j++) {
          const int k = kb0 + j * 16 + l16;
          float v = (k <= q) ? csr[j][r] * scale : NEG_BIG;
          csr[j][r] = v;
          rm[r] = fmaxf(rm[r], v);
        }
      }
#pragma unroll
      for (int r = 0; r < 8; r++) {
#pragma unroll
        for (int m = 8; m >= 1; m >>= 1) rm[r] = fmaxf(rm[r], __shfl_xor(rm[r], m, 32));
        float mn = fmaxf(m_i[r], rm[r]);
        float alpha = __expf(m_i[r] - mn);
        m_i[r] = mn;
        rs[r] = 0.f;
#pragma unroll
        for (int j = 0; j < 4; j++) {
          float p = __expf(csr[j][r] - mn);
          csr[j][r] = p;
          rs[r] += p;
        }
#pragma unroll
        for (int m = 8; m >= 1; m >>= 1) rs[r] += __shfl_xor(rs[r], m, 32);
        l_i[r] = l_i[r] * alpha + rs[r];
#pragma unroll
        for (int j = 0; j < 4; j++) o[j][r] *= alpha;
      }

      // Relayout P (C-layout) -> A-fragment layout via per-wave LDS strip
      _Float16* p0 = &sP[wave * 16 * 64];
#pragma unroll
      for (int r = 0; r < 8; r++) {
        const int prow = lhi ? 8 + r : r;
#pragma unroll
        for (int j = 0; j < 4; j++)
          p0[prow * 64 + j * 16 + l16] = (_Float16)csr[j][r];
      }
      // O += P * V  (K-dim = 64 keys -> two WMMA steps; B from sV [dim][key])
#pragma unroll
      for (int f = 0; f < 2; f++) {
        v16h pa;
        v8h lo = *(const v8h*)(p0 + l16 * 64 + f * 32 + lhi * 8);
        v8h hi = *(const v8h*)(p0 + l16 * 64 + f * 32 + lhi * 8 + 16);
#pragma unroll
        for (int e = 0; e < 8; e++) { pa[e] = lo[e]; pa[e + 8] = hi[e]; }
#pragma unroll
        for (int j = 0; j < 4; j++) {
          v16h bv;
          {
            const _Float16* q = &sV[cur][(j * 16 + l16) * 64 + f * 32 + lhi * 8];
            v8h blo = *(const v8h*)q;
            v8h bhi = *(const v8h*)(q + 16);
#pragma unroll
            for (int e = 0; e < 8; e++) { bv[e] = blo[e]; bv[e + 8] = bhi[e]; }
          }
          o[j] = __builtin_amdgcn_wmma_f32_16x16x32_f16(
              false, pa, false, bv, (short)0, o[j], false, false);
        }
      }
    }
    __syncthreads();
  }

  // normalize and emit f16 [B,T,H*64]
#pragma unroll
  for (int r = 0; r < 8; r++) {
    const int t = q0 + (lhi ? 8 + r : r);
    const float inv = 1.0f / l_i[r];
    _Float16* d = Oh + ((long)(b * SEQC + t)) * D_MODELC + h * D_KC;
#pragma unroll
    for (int j = 0; j < 4; j++)
      d[j * 16 + l16] = (_Float16)(o[j][r] * inv);
  }
}

// ---------------------------------------------------------------------------
// launch
// ---------------------------------------------------------------------------
extern "C" void kernel_launch(void* const* d_in, const int* in_sizes, int n_in,
                              void* d_out, int out_size, void* d_ws, size_t ws_size,
                              hipStream_t stream) {
  const float* x    = (const float*)d_in[0];   // [2,2048,1024]
  const float* qkvo = (const float*)d_in[1];   // [2560,1024]
  const float* qw   = (const float*)d_in[2];   // [64]
  const float* kw   = (const float*)d_in[3];   // [64]
  float* out = (float*)d_out;                  // [2,2048,1024] f32

  char* ws = (char*)d_ws;
  const size_t SZ_XB  = (size_t)MTOKC * D_MODELC * 2;          //  8,388,608
  const size_t SZ_WB  = (size_t)(QKV_SIZEC + D_MODELC) * D_MODELC * 2; // 5,242,880
  const size_t SZ_QKV = (size_t)MTOKC * QKV_SIZEC * 4;         // 25,165,824
  const size_t SZ_QH  = (size_t)MTOKC * D_MODELC * 2;          //  8,388,608
  const size_t SZ_KH  = (size_t)BATCHC * N_KVC * SEQC * D_KC * 2; // 2,097,152

  _Float16* xb  = (_Float16*)(ws);
  _Float16* wb  = (_Float16*)(ws + SZ_XB);
  float*    qkv = (float*)   (ws + SZ_XB + SZ_WB);
  _Float16* Qhp = (_Float16*)(ws + SZ_XB + SZ_WB + SZ_QKV);
  _Float16* Khp = (_Float16*)(ws + SZ_XB + SZ_WB + SZ_QKV + SZ_QH);
  _Float16* Vtp = (_Float16*)(ws + SZ_XB + SZ_WB + SZ_QKV + SZ_QH + SZ_KH);
  _Float16* Ahp = xb;  // reuse: x-f16 not needed after QKV GEMM

  // 1) pack to f16
  {
    int nx = MTOKC * D_MODELC;                       // 4,194,304
    int nw = (QKV_SIZEC + D_MODELC) * D_MODELC;      // 2,621,440
    cvt_f32_to_f16<<<(nx + 255) / 256, 256, 0, stream>>>(x, xb, nx);
    cvt_f32_to_f16<<<(nw + 255) / 256, 256, 0, stream>>>(qkvo, wb, nw);
  }
  // 2) QKV projection: [4096,1536] = xb @ wb[0:1536]^T
  gemm_f16_wmma<<<dim3(QKV_SIZEC / 128, MTOKC / 128), 256, 0, stream>>>(
      xb, wb, qkv, MTOKC, QKV_SIZEC, D_MODELC);
  // 3) RMSNorm + RoPE + layouts
  {
    int rows = BATCHC * SEQC * (N_HEADSC + 2 * N_KVC); // 98304
    norm_rope_kernel<<<rows / 8, 256, 0, stream>>>(qkv, qw, kw, Qhp, Khp, Vtp);
  }
  // 4) causal flash attention
  flash_attn_wmma<<<dim3(SEQC / 128, BATCHC * N_HEADSC), 256, 0, stream>>>(
      Qhp, Khp, Vtp, Ahp);
  // 5) output projection: out = Ah @ wb[1536:2560]^T (f32 out)
  gemm_f16_wmma<<<dim3(D_MODELC / 128, MTOKC / 128), 256, 0, stream>>>(
      Ahp, wb + (size_t)QKV_SIZEC * D_MODELC, out, MTOKC, D_MODELC, D_MODELC);
}
